// CIN_74603581932155
// MI455X (gfx1250) — matrix operand order repaired and checked by reading, not compiled, not run on hardware
//
#include <hip/hip_runtime.h>
#include <hip/hip_bf16.h>
#include <stdint.h>

typedef __attribute__((ext_vector_type(16))) __bf16 v16bf;
typedef __attribute__((ext_vector_type(8)))  float  v8f;

union Frag {
  uint32_t u[8];
  uint4    u4[2];
  v16bf    v;
};

static __device__ __forceinline__ uint16_t f32_to_bf16(float f) {
  uint32_t u = __float_as_uint(f);
  uint32_t r = u + 0x7FFFu + ((u >> 16) & 1u);  // round-to-nearest-even
  return (uint16_t)(r >> 16);
}

// Async copy of one 16B packet: global -> LDS (ASYNCcnt-tracked, CDNA5).
static __device__ __forceinline__ void async_b128(uint32_t lds_off, const void* g) {
  asm volatile("global_load_async_to_lds_b128 %0, %1, off"
               :: "v"(lds_off), "v"((uint64_t)(uintptr_t)g)
               : "memory");
}
static __device__ __forceinline__ void wait_async0() {
  asm volatile("s_wait_asynccnt 0x0" ::: "memory");
}

// ---------------------------------------------------------------------------
// Prepack W (fp32, [o, h*32+m]) into bf16 WMMA A-fragment order, CHUNK-MAJOR:
//   dst u32 index = ((h*8 + ot)*32 + L)*8 + v
// so one K-chunk (all 8 O-tiles) is a contiguous 8 KB burst for async staging.
// A-frag lane layout (16x32 bf16 A, ISA 7.12.2): lane L<16 row M=L holds
// K {0..7,16..23}; lane L>=16 row M=L-16 holds K {8..15,24..31}; pair=(k0,k0+1).
// ---------------------------------------------------------------------------
__global__ void cin_pack_w(const float* __restrict__ W, uint32_t* __restrict__ dst,
                           int Hprev) {
  int idx = blockIdx.x * blockDim.x + threadIdx.x;
  int total = 8 * Hprev * 32 * 8;
  if (idx >= total) return;
  int v   = idx & 7;
  int L   = (idx >> 3) & 31;
  int rem = idx >> 8;
  int ot  = rem & 7;
  int h   = rem >> 3;
  int K   = Hprev * 32;
  int o   = ot * 16 + (L & 15);
  int base_k = ((v < 4) ? 0 : 16) + ((L >= 16) ? 8 : 0);
  int k0  = base_k + 2 * (v & 3);
  const float* row = W + (size_t)o * K + (size_t)h * 32;
  uint32_t lo = f32_to_bf16(row[k0]);
  uint32_t hi = f32_to_bf16(row[k0 + 1]);
  dst[idx] = lo | (hi << 16);
}

__global__ void cin_cvt_x(const float* __restrict__ x, uint16_t* __restrict__ xbf,
                          int n) {
  int idx = blockIdx.x * blockDim.x + threadIdx.x;
  if (idx < n) xbf[idx] = f32_to_bf16(x[idx]);
}

// ---------------------------------------------------------------------------
// One CIN layer, fused scale-and-GEMM.
//   grid.x = B/2 (two batch elements per WG), 512 threads = 16 waves.
//   Wave (bw, ow): batch b0+bw, O-tile [ow*16, ow*16+16), all 4 d-tiles.
//   A chunks double-buffered in LDS via global_load_async_to_lds_b128;
//   B fragments built in-register: B[m,d] = x0[b,m,d] * hprev[b,h,d]
//   (v_pk_mul_bf16), accumulated with v_wmma_f32_16x16x32_bf16.
// ---------------------------------------------------------------------------
__global__ __launch_bounds__(512)
void cin_layer(const uint16_t* __restrict__ xbf,    // (B,32,64) bf16 base fields
               const uint16_t* __restrict__ hprev,  // (B,Hprev,64) bf16
               const uint32_t* __restrict__ Wpk,    // packed A frags, chunk-major
               const float*   __restrict__ bias,    // (128)
               uint16_t*      __restrict__ hout,    // (B,128,64) bf16
               float*         __restrict__ out,     // (B,384)
               int Hprev, int out_off) {
  __shared__ __align__(16) uint16_t hs_lds[2 * 128 * 64];  // 32 KB
  __shared__ __align__(16) uint32_t a_lds[2][2048];        // 2 x 8 KB

  const int tid  = threadIdx.x;
  const int bw   = tid >> 8;          // which batch element of the pair
  const int ow   = (tid >> 5) & 7;    // O-tile
  const int lane = tid & 31;
  const int col  = lane & 15;         // column within 16-wide d-tile
  const int hi16 = lane >> 4;
  const int b0   = blockIdx.x * 2;
  const int b    = b0 + bw;

  const uint32_t hs_off = (uint32_t)(uintptr_t)(void*)hs_lds;
  const uint32_t a_off0 = (uint32_t)(uintptr_t)(void*)a_lds[0];
  const char*    wbytes = (const char*)Wpk;

  // Prologue: async-stage hprev for both b's, and A chunk 0.
  {
    const char* hsrc = (const char*)(hprev + (size_t)b0 * Hprev * 64);
    const int n16 = Hprev * 16;  // (2*Hprev*64*2 bytes) / 16
    for (int i = tid; i < n16; i += 512)
      async_b128(hs_off + i * 16, hsrc + (size_t)i * 16);
    async_b128(a_off0 + tid * 16, wbytes + (size_t)tid * 16);
  }

  // Per-lane x0 fragments: constant across the entire K loop.
  // B-frag layout (32x16 bf16 B): lane L holds col N=L%16, K=(L>=16?8:0)+(v>=4?8:0)+2v{+1}.
  uint32_t xpk[4][8];
  {
    const uint16_t* xb = xbf + (size_t)b * 32 * 64;
    const int mbase = hi16 ? 16 : 0;
#pragma unroll
    for (int t = 0; t < 4; ++t) {
      const int d = t * 16 + col;
#pragma unroll
      for (int v = 0; v < 8; ++v) {
        const int m0 = (hi16 ? 8 : 0) + (v < 4 ? 0 : 8) + 2 * v;
        uint32_t lo = xb[(size_t)m0 * 64 + d];
        uint32_t hi = xb[(size_t)(m0 + 1) * 64 + d];
        xpk[t][v] = lo | (hi << 16);
      }
    }
  }

  v8f zerov = {};
  v8f acc[4];
#pragma unroll
  for (int t = 0; t < 4; ++t) acc[t] = zerov;

  const uint16_t* hsb = hs_lds + (size_t)bw * Hprev * 64;

  for (int h = 0; h < Hprev; ++h) {
    wait_async0();      // my slice of chunk h (and hprev, on h==0) has landed
    __syncthreads();    // everyone's slices have landed; iter h-1 reads done

    if (h + 1 < Hprev)  // prefetch next chunk into the other buffer
      async_b128(a_off0 + (uint32_t)((h + 1) & 1) * 8192u + tid * 16,
                 wbytes + (size_t)(h + 1) * 8192 + tid * 16);

    Frag A;
    const uint4* ap = (const uint4*)((const char*)a_lds[h & 1]) +
                      ((size_t)ow * 32 + lane) * 2;
    A.u4[0] = ap[0];
    A.u4[1] = ap[1];

    const uint16_t* hrow = hsb + h * 64;
#pragma unroll
    for (int t = 0; t < 4; ++t) {
      // broadcast scale hprev[h, d] duplicated into both bf16 halves
      uint32_t hp = (uint32_t)hrow[t * 16 + col] * 0x00010001u;
      Frag B;
#pragma unroll
      for (int v = 0; v < 8; ++v) {
        uint32_t r;
        asm("v_pk_mul_bf16 %0, %1, %2" : "=v"(r) : "v"(xpk[t][v]), "v"(hp));
        B.u[v] = r;
      }
      acc[t] = __builtin_amdgcn_wmma_f32_16x16x32_bf16(
          false, A.v, false, B.v, (short)0, acc[t], false, false);
    }
  }

  // Epilogue: bias + ReLU, store h (bf16), and d-sum into out.
  float rs[8];
  float bb[8];
#pragma unroll
  for (int r = 0; r < 8; ++r) {
    rs[r] = 0.0f;
    bb[r] = bias[ow * 16 + r + hi16 * 8];
  }
  uint16_t* hb = hout + (size_t)b * 128 * 64;
#pragma unroll
  for (int t = 0; t < 4; ++t) {
    const int d = t * 16 + col;
#pragma unroll
    for (int r = 0; r < 8; ++r) {
      const int row = ow * 16 + r + hi16 * 8;  // C/D layout: vgpr r, half-wave
      float z  = acc[t][r] + bb[r];
      float hv = fmaxf(z, 0.0f);
      hb[(size_t)row * 64 + d] = f32_to_bf16(hv);
      rs[r] += hv;
    }
  }
#pragma unroll
  for (int r = 0; r < 8; ++r) {
    float v = rs[r];
    v += __shfl_xor(v, 8, 16);
    v += __shfl_xor(v, 4, 16);
    v += __shfl_xor(v, 2, 16);
    v += __shfl_xor(v, 1, 16);
    if (col == 0) {
      const int row = ow * 16 + r + hi16 * 8;
      out[(size_t)b * 384 + out_off + row] = v;
    }
  }
}

extern "C" void kernel_launch(void* const* d_in, const int* in_sizes, int n_in,
                              void* d_out, int out_size, void* d_ws, size_t ws_size,
                              hipStream_t stream) {
  (void)in_sizes; (void)n_in; (void)out_size; (void)ws_size;
  const float* x  = (const float*)d_in[0];
  const float* W0 = (const float*)d_in[1];
  const float* b0 = (const float*)d_in[2];
  const float* W1 = (const float*)d_in[3];
  const float* b1 = (const float*)d_in[4];
  const float* W2 = (const float*)d_in[5];
  const float* b2 = (const float*)d_in[6];
  float* out = (float*)d_out;

  char* ws = (char*)d_ws;
  uint16_t* xbf = (uint16_t*)(ws);                                   // 4 MB
  uint32_t* Wp0 = (uint32_t*)(ws + ((size_t)4 << 20));               // 256 KB
  uint32_t* Wp1 = (uint32_t*)(ws + ((size_t)4 << 20) + (256u << 10));// 1 MB
  uint32_t* Wp2 = (uint32_t*)(ws + ((size_t)5 << 20) + (256u << 10));// 1 MB
  uint16_t* h1  = (uint16_t*)(ws + ((size_t)8 << 20));               // 16 MB
  uint16_t* h2  = (uint16_t*)(ws + ((size_t)24 << 20));              // 16 MB

  // Pre-pass: bf16-convert x, pack W into chunk-major WMMA A-fragment order.
  const int nx = 1024 * 32 * 64;
  cin_cvt_x<<<(nx + 255) / 256, 256, 0, stream>>>(x, xbf, nx);
  cin_pack_w<<<(8 * 32  * 32 * 8) / 256, 256, 0, stream>>>(W0, Wp0, 32);
  cin_pack_w<<<(8 * 128 * 32 * 8) / 256, 256, 0, stream>>>(W1, Wp1, 128);
  cin_pack_w<<<(8 * 128 * 32 * 8) / 256, 256, 0, stream>>>(W2, Wp2, 128);

  // Three fused layers (2 batch elements per WG); layer 2 reuses h1 as its
  // (unused downstream) h store.
  cin_layer<<<512, 512, 0, stream>>>(xbf, xbf, Wp0, b0, h1, out, 32, 0);
  cin_layer<<<512, 512, 0, stream>>>(xbf, h1,  Wp1, b1, h2, out, 128, 128);
  cin_layer<<<512, 512, 0, stream>>>(xbf, h2,  Wp2, b2, h1, out, 128, 256);
}
